// AttentionBlock_2723009265670
// MI455X (gfx1250) — matrix operand order repaired and checked by reading, not compiled, not run on hardware
//
#include <hip/hip_runtime.h>
#include <hip/hip_bf16.h>
#include <math.h>

typedef _Float16 half_t;
typedef __attribute__((ext_vector_type(16))) _Float16     v16h;
typedef __attribute__((ext_vector_type(8)))  float        v8f;
typedef __attribute__((ext_vector_type(4)))  unsigned int v4u;
typedef __attribute__((ext_vector_type(4)))  int          v4i;

union FragH { v16h h; v4u u[2]; };

#define WMMA_F32_F16(a, b, c) \
  __builtin_amdgcn_wmma_f32_16x16x32_f16(false, (a), false, (b), (short)0, (c), false, false)

// ---- CDNA5 async global->LDS copies (ASYNCcnt) ----
#if defined(__HIP_DEVICE_COMPILE__) && __has_builtin(__builtin_amdgcn_global_load_async_to_lds_b128)
#define HAVE_ASYNC_LDS 1
typedef __attribute__((address_space(1))) v4i gv4i;
typedef __attribute__((address_space(3))) v4i lv4i;
__device__ __forceinline__ void async_copy_b128(const void* g, void* l) {
  __builtin_amdgcn_global_load_async_to_lds_b128((gv4i*)(g), (lv4i*)(l), 0, 0);
}
__device__ __forceinline__ void wait_async0() {
#if __has_builtin(__builtin_amdgcn_s_wait_asynccnt)
  __builtin_amdgcn_s_wait_asynccnt(0);
#else
  asm volatile("s_wait_asynccnt 0x0" ::: "memory");
#endif
}
#else
#define HAVE_ASYNC_LDS 0
#endif

__device__ __forceinline__ void stage_b128(const void* g, void* l) {
#if HAVE_ASYNC_LDS
  async_copy_b128(g, l);
#else
  *reinterpret_cast<v4u*>(l) = *reinterpret_cast<const v4u*>(g);
#endif
}
__device__ __forceinline__ void stage_fence() {
#if HAVE_ASYNC_LDS
  wait_async0();
#endif
}

constexpr int Bn = 4;      // batch
constexpr int Cc = 512;    // channels
constexpr int Ln = 2048;   // sequence length
constexpr int Hh = 8;      // heads
constexpr int Dd = 64;     // head dim
constexpr int Gg = 8;      // groupnorm groups

// ---------------- workspace layout (bytes) ----------------
constexpr size_t OFF_STATS = 0;                                  // B*G*2 floats
constexpr size_t OFF_WQKV  = 1024;                               // 3C*C f16
constexpr size_t OFF_WPROJ = OFF_WQKV + (size_t)3*Cc*Cc*2;       // C*C f16
constexpr size_t OFF_Q     = OFF_WPROJ + (size_t)Cc*Cc*2;        // B*H*L*D f16
constexpr size_t OFF_K     = OFF_Q + (size_t)Bn*Hh*Ln*Dd*2;
constexpr size_t OFF_V     = OFF_K + (size_t)Bn*Hh*Ln*Dd*2;
constexpr size_t OFF_AO    = OFF_V + (size_t)Bn*Hh*Ln*Dd*2;      // B*C*L f16

// ---------------- kernel 1: groupnorm statistics ----------------
__global__ void gn_stats_kernel(const float* __restrict__ x, float* __restrict__ stats) {
  const int bg = blockIdx.x;
  const int b = bg / Gg, g = bg % Gg;
  const float* p = x + ((size_t)b * Cc + (size_t)g * (Cc / Gg)) * Ln;
  const int N = (Cc / Gg) * Ln;  // 131072
  float s = 0.f, s2 = 0.f;
  for (int i = threadIdx.x; i < N; i += blockDim.x) {
    float v = p[i];
    s += v; s2 += v * v;
  }
  __shared__ float sh[256], sh2[256];
  sh[threadIdx.x] = s; sh2[threadIdx.x] = s2;
  __syncthreads();
  for (int off = 128; off > 0; off >>= 1) {
    if ((int)threadIdx.x < off) {
      sh[threadIdx.x]  += sh[threadIdx.x + off];
      sh2[threadIdx.x] += sh2[threadIdx.x + off];
    }
    __syncthreads();
  }
  if (threadIdx.x == 0) {
    float mean = sh[0] / (float)N;
    float var  = sh2[0] / (float)N - mean * mean;
    stats[bg * 2 + 0] = mean;
    stats[bg * 2 + 1] = rsqrtf(var + 1e-5f);
  }
}

// ---------------- kernel 2: f32 -> f16 cast ----------------
__global__ void cast_f32_to_f16(const float* __restrict__ in, half_t* __restrict__ out, int n) {
  int i = blockIdx.x * blockDim.x + threadIdx.x;
  if (i < n) out[i] = (half_t)in[i];
}

// ---------------- kernel 3: fused groupnorm + QKV GEMM (WMMA) ----------------
// Block tile: 64 (o) x 128 (l); 8 waves in 2(o) x 4(l); 2x2 subtiles per wave.
__global__ void __launch_bounds__(256) qkv_gemm_kernel(
    const float* __restrict__ x, const float* __restrict__ stats,
    const float* __restrict__ nw, const float* __restrict__ nb,
    const half_t* __restrict__ w16, const float* __restrict__ bias,
    half_t* __restrict__ q16, half_t* __restrict__ k16, half_t* __restrict__ v16) {
  const int b  = blockIdx.z;
  const int o0 = blockIdx.y * 64;
  const int l0 = blockIdx.x * 128;
  const int tid = threadIdx.x, lane = tid & 31, wid = tid >> 5;
  const int wo = (wid & 1) * 32, wl = (wid >> 1) * 32;
  const int m = lane & 15, hf = lane >> 4;

  __shared__ __align__(16) half_t Asub[64 * 32];   // [o][c]
  __shared__ __align__(16) half_t Bsub[128 * 32];  // [l][c]

  v8f acc[2][2] = {};

  const int arow = tid >> 2, acol = (tid & 3) * 8;   // A: 64 x 32, one b128 each
  const int bc = tid >> 3, bl16 = (tid & 7) * 16;    // B: 32(c) x 128(l), 16 l's each

  for (int kk = 0; kk < Cc; kk += 32) {
    stage_b128(w16 + (size_t)(o0 + arow) * Cc + kk + acol, &Asub[arow * 32 + acol]);
    {
      const int c = kk + bc;
      const float mean = stats[(b * Gg + (c >> 6)) * 2 + 0];
      const float rstd = stats[(b * Gg + (c >> 6)) * 2 + 1];
      const float sw = nw[c] * rstd;
      const float sb = nb[c] - mean * sw;  // h = x*sw + sb
      const float4* gp =
          reinterpret_cast<const float4*>(x + ((size_t)b * Cc + c) * Ln + l0 + bl16);
#pragma unroll
      for (int j = 0; j < 4; ++j) {
        float4 v = gp[j];
        const int lb = bl16 + j * 4;
        Bsub[(lb + 0) * 32 + bc] = (half_t)(v.x * sw + sb);
        Bsub[(lb + 1) * 32 + bc] = (half_t)(v.y * sw + sb);
        Bsub[(lb + 2) * 32 + bc] = (half_t)(v.z * sw + sb);
        Bsub[(lb + 3) * 32 + bc] = (half_t)(v.w * sw + sb);
      }
    }
    stage_fence();
    __syncthreads();

    FragH a[2], bf[2];
#pragma unroll
    for (int i = 0; i < 2; ++i) {
      const v4u* ap = reinterpret_cast<const v4u*>(&Asub[(wo + i * 16 + m) * 32 + hf * 16]);
      a[i].u[0] = ap[0]; a[i].u[1] = ap[1];
      const v4u* bp = reinterpret_cast<const v4u*>(&Bsub[(wl + i * 16 + m) * 32 + hf * 16]);
      bf[i].u[0] = bp[0]; bf[i].u[1] = bp[1];
    }
#pragma unroll
    for (int ti = 0; ti < 2; ++ti)
#pragma unroll
      for (int tj = 0; tj < 2; ++tj)
        acc[ti][tj] = WMMA_F32_F16(a[ti].h, bf[tj].h, acc[ti][tj]);
    __syncthreads();
  }

  // epilogue: route to q/k/v in [b,h,l,d]; q pre-scaled by D^-0.5
#pragma unroll
  for (int ti = 0; ti < 2; ++ti) {
#pragma unroll
    for (int tj = 0; tj < 2; ++tj) {
      const int l = l0 + wl + tj * 16 + m;
#pragma unroll
      for (int r = 0; r < 8; ++r) {
        int o = o0 + wo + ti * 16 + r + 8 * hf;
        float val = acc[ti][tj][r] + bias[o];
        int which = o >> 9;       // 0=q, 1=k, 2=v
        int oc = o & 511;
        int head = oc >> 6, d = oc & 63;
        size_t idx = (((size_t)b * Hh + head) * Ln + l) * Dd + d;
        if (which == 0)       q16[idx] = (half_t)(val * 0.125f);
        else if (which == 1)  k16[idx] = (half_t)val;
        else                  v16[idx] = (half_t)val;
      }
    }
  }
}

// ---------------- kernel 4: flash attention (WMMA + online softmax) ----------------
// Block: 128 query rows (8 waves x 16 rows); loop over 32-key tiles.
__global__ void __launch_bounds__(256) attn_kernel(
    const half_t* __restrict__ q16, const half_t* __restrict__ k16,
    const half_t* __restrict__ v16, half_t* __restrict__ ao16) {
  const int b = blockIdx.z, hd = blockIdx.y;
  const int tid = threadIdx.x, lane = tid & 31, wid = tid >> 5;
  const int m = lane & 15, hf = lane >> 4;
  const int lq0 = blockIdx.x * 128 + wid * 16;

  __shared__ __align__(16) half_t Klds[32 * 64];    // [key][d]
  __shared__ __align__(16) half_t VTlds[64 * 32];   // [d][key]
  __shared__ __align__(16) half_t Plds[8][16 * 32]; // per-wave [row][key]

  const size_t headoff = ((size_t)b * Hh + hd) * Ln;

  // resident Q fragments (d-chunks [0,32) and [32,64)); q already pre-scaled
  FragH qa0, qa1;
  {
    const v4u* gp = reinterpret_cast<const v4u*>(q16 + (headoff + lq0 + m) * Dd);
    qa0.u[0] = gp[0 + 2 * hf];  // d = hf*16
    qa0.u[1] = gp[1 + 2 * hf];  // d = 8 + hf*16
    qa1.u[0] = gp[4 + 2 * hf];  // d = 32 + hf*16
    qa1.u[1] = gp[5 + 2 * hf];  // d = 40 + hf*16
  }

  float ms[8], ls[8];
  v8f zero = {};
  v8f oacc[4];
#pragma unroll
  for (int r = 0; r < 8; ++r) { ms[r] = -3.0e38f; ls[r] = 0.f; }
#pragma unroll
  for (int t = 0; t < 4; ++t) oacc[t] = zero;

  const int key = tid >> 3, d8 = (tid & 7) * 8;  // staging coords (32 keys x 64 d)

  for (int j0 = 0; j0 < Ln; j0 += 32) {
    // stage K tile via async copy; V tile transposed via VALU
    stage_b128(k16 + (headoff + j0 + key) * Dd + d8, &Klds[key * 64 + d8]);
    {
      const half_t* vp = v16 + (headoff + j0 + key) * Dd + d8;
#pragma unroll
      for (int i = 0; i < 8; ++i) VTlds[(d8 + i) * 32 + key] = vp[i];
    }
    if (j0 + 32 < Ln) {  // cover next tile's latency
      __builtin_prefetch(k16 + (headoff + j0 + 32 + key) * Dd + d8, 0, 0);
      __builtin_prefetch(v16 + (headoff + j0 + 32 + key) * Dd + d8, 0, 0);
    }
    stage_fence();
    __syncthreads();

    // scores: S[16 x 32] as two 16x16 tiles, K = d accumulated over 2 chunks
    v8f s0 = zero, s1 = zero;
    {
      FragH bk;
      const v4u* p0 = reinterpret_cast<const v4u*>(&Klds[m * 64 + hf * 16]);
      bk.u[0] = p0[0]; bk.u[1] = p0[1];
      s0 = WMMA_F32_F16(qa0.h, bk.h, s0);
      const v4u* p1 = reinterpret_cast<const v4u*>(&Klds[m * 64 + 32 + hf * 16]);
      bk.u[0] = p1[0]; bk.u[1] = p1[1];
      s0 = WMMA_F32_F16(qa1.h, bk.h, s0);
      const v4u* p2 = reinterpret_cast<const v4u*>(&Klds[(16 + m) * 64 + hf * 16]);
      bk.u[0] = p2[0]; bk.u[1] = p2[1];
      s1 = WMMA_F32_F16(qa0.h, bk.h, s1);
      const v4u* p3 = reinterpret_cast<const v4u*>(&Klds[(16 + m) * 64 + 32 + hf * 16]);
      bk.u[0] = p3[0]; bk.u[1] = p3[1];
      s1 = WMMA_F32_F16(qa1.h, bk.h, s1);
    }

    // online softmax; row r's 16 columns live in one half-wave -> shfl width 16
#pragma unroll
    for (int r = 0; r < 8; ++r) {
      float t = fmaxf(s0[r], s1[r]);
#pragma unroll
      for (int off = 8; off > 0; off >>= 1) t = fmaxf(t, __shfl_xor(t, off, 16));
      float mn = fmaxf(ms[r], t);
      float alpha = __expf(ms[r] - mn);
      float p0 = __expf(s0[r] - mn);
      float p1 = __expf(s1[r] - mn);
      float rs = p0 + p1;
#pragma unroll
      for (int off = 8; off > 0; off >>= 1) rs += __shfl_xor(rs, off, 16);
      ls[r] = ls[r] * alpha + rs;
      ms[r] = mn;
      int row = r + 8 * hf;
      Plds[wid][row * 32 + m]      = (half_t)p0;
      Plds[wid][row * 32 + 16 + m] = (half_t)p1;
#pragma unroll
      for (int t4 = 0; t4 < 4; ++t4) oacc[t4][r] *= alpha;
    }

    // O += P(16x32) x V(32x64): wave-private P, no extra barrier needed
    FragH pa;
    const v4u* pp = reinterpret_cast<const v4u*>(&Plds[wid][m * 32 + hf * 16]);
    pa.u[0] = pp[0];
    pa.u[1] = pp[1];
#pragma unroll
    for (int t4 = 0; t4 < 4; ++t4) {
      FragH bv;
      const v4u* vp = reinterpret_cast<const v4u*>(&VTlds[(t4 * 16 + m) * 32 + hf * 16]);
      bv.u[0] = vp[0]; bv.u[1] = vp[1];
      oacc[t4] = WMMA_F32_F16(pa.h, bv.h, oacc[t4]);
    }
    __syncthreads();
  }

  // epilogue: normalize by row sum, store transposed into [b, c, l] for proj GEMM
#pragma unroll
  for (int t4 = 0; t4 < 4; ++t4) {
    int d = t4 * 16 + m;
    int c = hd * Dd + d;
#pragma unroll
    for (int r = 0; r < 8; ++r) {
      int lq = lq0 + r + 8 * hf;
      float val = oacc[t4][r] / ls[r];
      ao16[((size_t)b * Cc + c) * Ln + lq] = (half_t)val;
    }
  }
}

// ---------------- kernel 5: projection GEMM + bias + residual (WMMA) ----------------
__global__ void __launch_bounds__(256) proj_gemm_kernel(
    const half_t* __restrict__ ao16, const half_t* __restrict__ w16,
    const float* __restrict__ bias, const float* __restrict__ x,
    float* __restrict__ out) {
  const int b  = blockIdx.z;
  const int o0 = blockIdx.y * 64;
  const int l0 = blockIdx.x * 128;
  const int tid = threadIdx.x, lane = tid & 31, wid = tid >> 5;
  const int wo = (wid & 1) * 32, wl = (wid >> 1) * 32;
  const int m = lane & 15, hf = lane >> 4;

  __shared__ __align__(16) half_t Asub[64 * 32];
  __shared__ __align__(16) half_t Bsub[128 * 32];

  v8f acc[2][2] = {};

  const int arow = tid >> 2, acol = (tid & 3) * 8;
  const int bc = tid >> 3, bl16 = (tid & 7) * 16;

  for (int kk = 0; kk < Cc; kk += 32) {
    stage_b128(w16 + (size_t)(o0 + arow) * Cc + kk + acol, &Asub[arow * 32 + acol]);
    {
      const int c = kk + bc;
      const half_t* gp = ao16 + ((size_t)b * Cc + c) * Ln + l0 + bl16;
#pragma unroll
      for (int j = 0; j < 16; ++j) Bsub[(bl16 + j) * 32 + bc] = gp[j];
    }
    stage_fence();
    __syncthreads();

    FragH a[2], bf[2];
#pragma unroll
    for (int i = 0; i < 2; ++i) {
      const v4u* ap = reinterpret_cast<const v4u*>(&Asub[(wo + i * 16 + m) * 32 + hf * 16]);
      a[i].u[0] = ap[0]; a[i].u[1] = ap[1];
      const v4u* bp = reinterpret_cast<const v4u*>(&Bsub[(wl + i * 16 + m) * 32 + hf * 16]);
      bf[i].u[0] = bp[0]; bf[i].u[1] = bp[1];
    }
#pragma unroll
    for (int ti = 0; ti < 2; ++ti)
#pragma unroll
      for (int tj = 0; tj < 2; ++tj)
        acc[ti][tj] = WMMA_F32_F16(a[ti].h, bf[tj].h, acc[ti][tj]);
    __syncthreads();
  }

#pragma unroll
  for (int ti = 0; ti < 2; ++ti) {
#pragma unroll
    for (int tj = 0; tj < 2; ++tj) {
      const int l = l0 + wl + tj * 16 + m;
#pragma unroll
      for (int r = 0; r < 8; ++r) {
        int o = o0 + wo + ti * 16 + r + 8 * hf;
        size_t idx = ((size_t)b * Cc + o) * Ln + l;
        out[idx] = x[idx] + bias[o] + acc[ti][tj][r];
      }
    }
  }
}

// ---------------- launch ----------------
extern "C" void kernel_launch(void* const* d_in, const int* in_sizes, int n_in,
                              void* d_out, int out_size, void* d_ws, size_t ws_size,
                              hipStream_t stream) {
  const float* x      = (const float*)d_in[0];
  const float* norm_w = (const float*)d_in[1];
  const float* norm_b = (const float*)d_in[2];
  const float* qkv_w  = (const float*)d_in[3];
  const float* qkv_b  = (const float*)d_in[4];
  const float* proj_w = (const float*)d_in[5];
  const float* proj_b = (const float*)d_in[6];
  float* out = (float*)d_out;

  char* ws = (char*)d_ws;
  float*  stats   = (float*)(ws + OFF_STATS);
  half_t* wqkv16  = (half_t*)(ws + OFF_WQKV);
  half_t* wproj16 = (half_t*)(ws + OFF_WPROJ);
  half_t* q16     = (half_t*)(ws + OFF_Q);
  half_t* k16     = (half_t*)(ws + OFF_K);
  half_t* v16     = (half_t*)(ws + OFF_V);
  half_t* ao16    = (half_t*)(ws + OFF_AO);

  gn_stats_kernel<<<Bn * Gg, 256, 0, stream>>>(x, stats);

  int nqkvw = 3 * Cc * Cc;
  cast_f32_to_f16<<<(nqkvw + 255) / 256, 256, 0, stream>>>(qkv_w, wqkv16, nqkvw);
  int nprojw = Cc * Cc;
  cast_f32_to_f16<<<(nprojw + 255) / 256, 256, 0, stream>>>(proj_w, wproj16, nprojw);

  qkv_gemm_kernel<<<dim3(Ln / 128, (3 * Cc) / 64, Bn), 256, 0, stream>>>(
      x, stats, norm_w, norm_b, wqkv16, qkv_b, q16, k16, v16);

  attn_kernel<<<dim3(Ln / 128, Hh, Bn), 256, 0, stream>>>(q16, k16, v16, ao16);

  proj_gemm_kernel<<<dim3(Ln / 128, Cc / 64, Bn), 256, 0, stream>>>(
      ao16, wproj16, proj_b, x, out);
}
